// MultiHeadAttention_72026601554422
// MI455X (gfx1250) — matrix operand order repaired and checked by reading, not compiled, not run on hardware
//
#include <hip/hip_runtime.h>
#include <cstdint>
#include <cstddef>

// ---------------------------------------------------------------------------
// Types matching the gfx1250 WMMA / TDM builtin signatures
// ---------------------------------------------------------------------------
typedef __attribute__((ext_vector_type(16))) __bf16 bf16x16;
typedef __attribute__((ext_vector_type(8)))  __bf16 bf16x8;
typedef __attribute__((ext_vector_type(4)))  __bf16 bf16x4;
typedef __attribute__((ext_vector_type(8)))  float  f32x8;
typedef __attribute__((ext_vector_type(4)))  float  f32x4;
typedef __attribute__((ext_vector_type(4)))  unsigned int u32x4;
typedef __attribute__((ext_vector_type(8)))  int    i32x8;
typedef __attribute__((ext_vector_type(4)))  int    i32x4;

// Build a 16-element bf16 operand from two 16-byte (8-elem) LDS chunks.
// Emits two ds_load_b128 when pointers are 16B aligned (our LDS strides are).
__device__ __forceinline__ bf16x16 ld_frag(const __bf16* lo, const __bf16* hi) {
  bf16x16 r;
  ((bf16x8*)&r)[0] = *(const bf16x8*)lo;
  ((bf16x8*)&r)[1] = *(const bf16x8*)hi;
  return r;
}

__device__ __forceinline__ f32x8 wmma_bf16(bf16x16 a, bf16x16 b, f32x8 c) {
  // (neg_a, A, neg_b, B, c_mod, C, reuse_a, reuse_b)
  return __builtin_amdgcn_wmma_f32_16x16x32_bf16(false, a, false, b, (short)0, c,
                                                 false, false);
}

__device__ __forceinline__ bf16x4 cvt4(f32x4 f) {
  bf16x4 h;
  h[0] = (__bf16)f[0]; h[1] = (__bf16)f[1];
  h[2] = (__bf16)f[2]; h[3] = (__bf16)f[3];
  return h;
}

// ---------------------------------------------------------------------------
// TDM: DMA a 128-row x 64-elem (bf16) tile from global to LDS, with hardware
// row padding: 128B of data then 16B pad  =>  LDS row stride 144B (LD = 72).
// D# packing per CDNA5 ISA:
//   group0: count=1 | lds_addr | global_addr[56:0] | type=2
//   group1: data_size=2B, pad_enable, pad_interval=4 (32 DW = 128B),
//           pad_amount=3 (4 DW = 16B), tensor_dim0=64, tensor_dim1=128,
//           tile_dim0=64, tile_dim1=128, tensor_dim0_stride=1024
//   groups 2/3: zero (2-D tile)
// This toolchain exposes the 6-arg builtin (extra int32x8 group before cpol).
// ---------------------------------------------------------------------------
__device__ __forceinline__ void tdm_load_k_tile(const __bf16* gsrc,
                                                unsigned lds_addr) {
  unsigned long long ga = (unsigned long long)(uintptr_t)gsrc;
  u32x4 g0;
  g0[0] = 1u;                                            // count=1, user mode
  g0[1] = lds_addr;                                      // LDS byte address
  g0[2] = (unsigned)(ga & 0xFFFFFFFFu);                  // global_addr[31:0]
  g0[3] = (unsigned)((ga >> 32) & 0x1FFFFFFu) | (2u << 30);  // [56:32] | type=2
  i32x8 g1;
  g1[0] = (int)((1u << 16) | (1u << 20) | (4u << 22) | (3u << 25));
  g1[1] = (int)(64u << 16);    // tensor_dim0 = 64   (bits 79:48, low part)
  g1[2] = (int)(128u << 16);   // tensor_dim1 = 128  (bits 111:80, low part)
  g1[3] = (int)(64u << 16);    // tile_dim0 = 64     (bits 127:112)
  g1[4] = (int)(128u);         // tile_dim1 = 128, tile_dim2 = 0
  g1[5] = (int)(1024u);        // tensor_dim0_stride = 1024 elems (48b, low)
  g1[6] = 0;
  g1[7] = 0;
  i32x4 z4 = {0, 0, 0, 0};
  i32x8 z8 = {0, 0, 0, 0, 0, 0, 0, 0};
  __builtin_amdgcn_tensor_load_to_lds(g0, g1, z4, z4, z8, 0);
}

// Problem constants
#define BATCH 8
#define SEQ   1024
#define DMODEL 1024
#define NHEAD 16
#define DEPTH 64

// ---------------------------------------------------------------------------
// Kernel 1/4: C[M,1024] = A[M,1024] @ W[1024,1024] + bias   (M = 8192)
// A fp32 or bf16; C bf16 (workspace) or fp32 (final output).
// Block: 256 thr = 8 waves; block tile 128x128; wave tile 32x64; BK=32.
// ---------------------------------------------------------------------------
template<bool A_BF16, bool OUT_F32>
__global__ __launch_bounds__(256)
void gemm_bias_kernel(const void* __restrict__ Ain, const float* __restrict__ W,
                      const float* __restrict__ bias, void* __restrict__ Cout) {
  constexpr int Kdim = 1024, Ndim = 1024;
  constexpr int BM = 128, BN = 128, BK = 32;
  constexpr int LDA = 48, LDB = 48;  // bf16 elems; row = 96 B, 16B-aligned

  __shared__ __align__(16) __bf16 As[BM * LDA];   // row-major [m][k]
  __shared__ __align__(16) __bf16 Bt[BN * LDB];   // N-major   [n][k]

  const int tid  = threadIdx.x;
  const int lane = tid & 31, wave = tid >> 5;
  const int wm = wave >> 1, wn = wave & 1;        // 4x2 wave grid
  const int lm = lane & 15, lh = lane >> 4;
  const int m0 = blockIdx.y * BM, n0 = blockIdx.x * BN;

  f32x8 acc[2][4];
  const f32x8 zero = {0.f, 0.f, 0.f, 0.f, 0.f, 0.f, 0.f, 0.f};
#pragma unroll
  for (int mi = 0; mi < 2; ++mi)
#pragma unroll
    for (int ni = 0; ni < 4; ++ni) acc[mi][ni] = zero;

  for (int k0 = 0; k0 < Kdim; k0 += BK) {
    // Stage A tile (convert to bf16 if needed): 128x32, 4 chunks/thread
    if (A_BF16) {
      const __bf16* A = (const __bf16*)Ain;
      for (int i = tid; i < BM * BK / 4; i += 256) {
        int r = i >> 3, c = (i & 7) * 4;
        *(bf16x4*)&As[r * LDA + c] =
            *(const bf16x4*)(A + (size_t)(m0 + r) * Kdim + k0 + c);
      }
    } else {
      const float* A = (const float*)Ain;
      for (int i = tid; i < BM * BK / 4; i += 256) {
        int r = i >> 3, c = (i & 7) * 4;
        f32x4 f = *(const f32x4*)(A + (size_t)(m0 + r) * Kdim + k0 + c);
        *(bf16x4*)&As[r * LDA + c] = cvt4(f);
      }
    }
    // Stage W tile transposed: Bt[n][k] = W[k0+k][n0+n] (coalesced global read)
    for (int i = tid; i < BK * BN; i += 256) {
      int kk = i >> 7, nn = i & 127;
      Bt[nn * LDB + kk] = (__bf16)W[(size_t)(k0 + kk) * Ndim + n0 + nn];
    }
    // Prefetch next K-tile (global_prefetch_b8; no counter cost)
    if (k0 + BK < Kdim && tid < 128) {
      if (A_BF16)
        __builtin_prefetch((const __bf16*)Ain + (size_t)(m0 + tid) * Kdim + k0 + BK, 0, 1);
      else
        __builtin_prefetch((const float*)Ain + (size_t)(m0 + tid) * Kdim + k0 + BK, 0, 1);
      int kk = tid >> 2, seg = tid & 3;
      __builtin_prefetch(W + (size_t)(k0 + BK + kk) * Ndim + n0 + seg * 32, 0, 1);
    }
    __syncthreads();

    bf16x16 af[2], bfrag[4];
#pragma unroll
    for (int mi = 0; mi < 2; ++mi) {
      const __bf16* base = &As[(wm * 32 + mi * 16 + lm) * LDA];
      af[mi] = ld_frag(base + lh * 8, base + 16 + lh * 8);
    }
#pragma unroll
    for (int ni = 0; ni < 4; ++ni) {
      const __bf16* base = &Bt[(wn * 64 + ni * 16 + lm) * LDB];
      bfrag[ni] = ld_frag(base + lh * 8, base + 16 + lh * 8);
    }
#pragma unroll
    for (int mi = 0; mi < 2; ++mi)
#pragma unroll
      for (int ni = 0; ni < 4; ++ni)
        acc[mi][ni] = wmma_bf16(af[mi], bfrag[ni], acc[mi][ni]);
    __syncthreads();
  }

  // Epilogue: +bias, store (C/D layout: M = r + 8*lh, N = lm per VGPR r)
#pragma unroll
  for (int ni = 0; ni < 4; ++ni) {
    int col = n0 + wn * 64 + ni * 16 + lm;
    float bv = bias[col];
#pragma unroll
    for (int mi = 0; mi < 2; ++mi) {
      int rbase = m0 + wm * 32 + mi * 16 + lh * 8;
#pragma unroll
      for (int r = 0; r < 8; ++r) {
        float v = acc[mi][ni][r] + bv;
        size_t idx = (size_t)(rbase + r) * Ndim + col;
        if (OUT_F32) ((float*)Cout)[idx] = v;
        else         ((__bf16*)Cout)[idx] = (__bf16)v;
      }
    }
  }
}

// ---------------------------------------------------------------------------
// Kernel 2: FUSED scores + softmax, TDM-fed.
// One block owns 64 q-rows x ALL 1024 k-cols of one (b,h).
// K panel streamed in 8 chunks of 128 rows via tensor_load_to_lds into a
// DOUBLE-BUFFERED padded LDS tile (hardware pad reproduces LD=72 layout);
// wave 0 issues chunk c+1 before waiting on chunk c (s_wait_tensorcnt 1),
// overlapping DMA with the WMMA phase. Row softmax entirely in registers
// (shuffle-xor across the 16 lanes of a row + LDS exchange across 2 n-waves);
// the NORMALIZED attn is written exactly once.
// ---------------------------------------------------------------------------
__global__ __launch_bounds__(256)
void scores_softmax_kernel(const __bf16* __restrict__ qb,
                           const __bf16* __restrict__ kb,
                           const float* __restrict__ mask,
                           float* __restrict__ attn) {
  constexpr int LD = 72;  // 144 B rows = 128B data + 16B TDM pad
  __shared__ __align__(16) __bf16 Qs[64 * LD];
  __shared__ __align__(16) __bf16 Ks[2][128 * LD];
  __shared__ float redmax[64][2];
  __shared__ float redsum[64][2];

  const int bh = blockIdx.y, b = bh >> 4, h = bh & 15;
  const int q0 = blockIdx.x * 64;
  const int tid = threadIdx.x, lane = tid & 31, wave = tid >> 5;
  const int wm = wave >> 1, wn = wave & 1, lm = lane & 15, lh = lane >> 4;

  const __bf16* kpanel = kb + (size_t)b * SEQ * DMODEL + h * DEPTH;

  // Stage the 64x64 Q strip (overlaps with first TDM chunk below)
  for (int i = tid; i < 64 * 16; i += 256) {
    int r = i >> 4, c = (i & 15) * 4;
    *(bf16x4*)&Qs[r * LD + c] =
        *(const bf16x4*)(qb + ((size_t)(b * SEQ + q0 + r)) * DMODEL + h * DEPTH + c);
  }
  if (wave == 0)
    tdm_load_k_tile(kpanel, (unsigned)(uintptr_t)&Ks[0][0]);
  __syncthreads();   // Qs ready (Ks[0] still in flight; gated per-chunk below)

  // A-fragments are invariant across all K chunks: preload both k-steps.
  bf16x16 af[2];
#pragma unroll
  for (int ks = 0; ks < 2; ++ks) {
    const __bf16* base = &Qs[(wm * 16 + lm) * LD + ks * 32];
    af[ks] = ld_frag(base + lh * 8, base + 16 + lh * 8);
  }

  f32x8 acc[32];
  const f32x8 zero = {0.f, 0.f, 0.f, 0.f, 0.f, 0.f, 0.f, 0.f};
#pragma unroll
  for (int f = 0; f < 32; ++f) acc[f] = zero;

#pragma unroll
  for (int c = 0; c < 8; ++c) {     // 8 chunks of 128 k-rows, double-buffered
    if (wave == 0) {
      if (c < 7) {
        tdm_load_k_tile(kpanel + (size_t)(c + 1) * 128 * DMODEL,
                        (unsigned)(uintptr_t)&Ks[(c + 1) & 1][0]);
        __builtin_amdgcn_s_wait_tensorcnt(1);   // chunk c complete
      } else {
        __builtin_amdgcn_s_wait_tensorcnt(0);
      }
    }
    __syncthreads();                // Ks[c&1] visible to all waves
#pragma unroll
    for (int ks = 0; ks < 2; ++ks) {
#pragma unroll
      for (int j = 0; j < 4; ++j) {
        const __bf16* base = &Ks[c & 1][(wn * 64 + j * 16 + lm) * LD + ks * 32];
        bf16x16 bfrag = ld_frag(base + lh * 8, base + 16 + lh * 8);
        acc[c * 4 + j] = wmma_bf16(af[ks], bfrag, acc[c * 4 + j]);
      }
    }
    __syncthreads();                // done with Ks[c&1] before it is re-DMAed
  }

  // ---- scale + mask + row max (registers) ----
  const int rowl = wm * 16 + lh * 8;   // + r  => local row id (0..63)
  float rmax[8];
#pragma unroll
  for (int r = 0; r < 8; ++r) rmax[r] = -3.0e38f;
#pragma unroll
  for (int f = 0; f < 32; ++f) {
    int col = (f >> 2) * 128 + wn * 64 + (f & 3) * 16 + lm;
    float mval = mask[b * SEQ + col] * (-1e9f);
#pragma unroll
    for (int r = 0; r < 8; ++r) {
      float v = acc[f][r] * 0.125f + mval;
      acc[f][r] = v;
      rmax[r] = fmaxf(rmax[r], v);
    }
  }
#pragma unroll
  for (int msk = 1; msk < 16; msk <<= 1)
#pragma unroll
    for (int r = 0; r < 8; ++r)
      rmax[r] = fmaxf(rmax[r], __shfl_xor(rmax[r], msk, 32));
  if (lm == 0) {
#pragma unroll
    for (int r = 0; r < 8; ++r) redmax[rowl + r][wn] = rmax[r];
  }
  __syncthreads();
#pragma unroll
  for (int r = 0; r < 8; ++r) rmax[r] = fmaxf(rmax[r], redmax[rowl + r][wn ^ 1]);

  // ---- exp + row sum ----
  float rsum[8];
#pragma unroll
  for (int r = 0; r < 8; ++r) rsum[r] = 0.f;
#pragma unroll
  for (int f = 0; f < 32; ++f)
#pragma unroll
    for (int r = 0; r < 8; ++r) {
      float e = __expf(acc[f][r] - rmax[r]);
      acc[f][r] = e;
      rsum[r] += e;
    }
#pragma unroll
  for (int msk = 1; msk < 16; msk <<= 1)
#pragma unroll
    for (int r = 0; r < 8; ++r) rsum[r] += __shfl_xor(rsum[r], msk, 32);
  if (lm == 0) {
#pragma unroll
    for (int r = 0; r < 8; ++r) redsum[rowl + r][wn] = rsum[r];
  }
  __syncthreads();
#pragma unroll
  for (int r = 0; r < 8; ++r) {
    rsum[r] += redsum[rowl + r][wn ^ 1];
    rsum[r] = 1.0f / rsum[r];
  }

  // ---- single normalized store ----
  float* arow = attn + ((size_t)bh * SEQ + q0) * SEQ;
#pragma unroll
  for (int f = 0; f < 32; ++f) {
    int col = (f >> 2) * 128 + wn * 64 + (f & 3) * 16 + lm;
#pragma unroll
    for (int r = 0; r < 8; ++r)
      arow[(size_t)(rowl + r) * SEQ + col] = acc[f][r] * rsum[r];
  }
}

// ---------------------------------------------------------------------------
// Kernel 3: ctx[b,s,h*64+d] = attn[b,h,s,:] @ v[b,:,h*64+d]   (bf16 out)
// Block tile 128(m) x 64(n); K loop over SEQ in steps of 32.
// ---------------------------------------------------------------------------
__global__ __launch_bounds__(256)
void attnv_kernel(const float* __restrict__ attn, const __bf16* __restrict__ vb,
                  __bf16* __restrict__ ctx) {
  constexpr int LD = 48;
  __shared__ __align__(16) __bf16 As[128 * LD];  // attn tile (bf16)
  __shared__ __align__(16) __bf16 Vt[64 * LD];   // v^T tile: [n][k]

  const int bh = blockIdx.y, b = bh >> 4, h = bh & 15;
  const int m0 = blockIdx.x * 128;
  const int tid = threadIdx.x, lane = tid & 31, wave = tid >> 5;
  const int lm = lane & 15, lh = lane >> 4;
  const float* arow = attn + (size_t)bh * SEQ * SEQ;

  f32x8 acc[4];
  const f32x8 zero = {0.f, 0.f, 0.f, 0.f, 0.f, 0.f, 0.f, 0.f};
#pragma unroll
  for (int ni = 0; ni < 4; ++ni) acc[ni] = zero;

  for (int k0 = 0; k0 < SEQ; k0 += 32) {
    for (int i = tid; i < 128 * 8; i += 256) {   // 128 rows x 32 cols / 4
      int r = i >> 3, c = (i & 7) * 4;
      f32x4 f = *(const f32x4*)(arow + (size_t)(m0 + r) * SEQ + k0 + c);
      *(bf16x4*)&As[r * LD + c] = cvt4(f);
    }
    for (int i = tid; i < 32 * 64; i += 256) {   // transpose v rows into Vt
      int kk = i >> 6, nn = i & 63;
      Vt[nn * LD + kk] =
          vb[((size_t)(b * SEQ + k0 + kk)) * DMODEL + h * DEPTH + nn];
    }
    if (k0 + 32 < SEQ && tid < 128) {
      __builtin_prefetch(arow + (size_t)(m0 + tid) * SEQ + k0 + 32, 0, 1);
      if (tid < 32)
        __builtin_prefetch(vb + ((size_t)(b * SEQ + k0 + 32 + tid)) * DMODEL + h * DEPTH,
                           0, 1);
    }
    __syncthreads();

    const __bf16* abase = &As[(wave * 16 + lm) * LD];
    bf16x16 af = ld_frag(abase + lh * 8, abase + 16 + lh * 8);
#pragma unroll
    for (int ni = 0; ni < 4; ++ni) {
      const __bf16* base = &Vt[(ni * 16 + lm) * LD];
      bf16x16 bfrag = ld_frag(base + lh * 8, base + 16 + lh * 8);
      acc[ni] = wmma_bf16(af, bfrag, acc[ni]);
    }
    __syncthreads();
  }

#pragma unroll
  for (int ni = 0; ni < 4; ++ni) {
    int col = ni * 16 + lm;
    int rbase = m0 + wave * 16 + lh * 8;
#pragma unroll
    for (int r = 0; r < 8; ++r)
      ctx[((size_t)(b * SEQ + rbase + r)) * DMODEL + h * DEPTH + col] =
          (__bf16)acc[ni][r];
  }
}

// ---------------------------------------------------------------------------
// Host launcher
// ---------------------------------------------------------------------------
extern "C" void kernel_launch(void* const* d_in, const int* in_sizes, int n_in,
                              void* d_out, int out_size, void* d_ws, size_t ws_size,
                              hipStream_t stream) {
  (void)in_sizes; (void)n_in; (void)out_size; (void)ws_size;
  const float* Q    = (const float*)d_in[0];
  const float* K    = (const float*)d_in[1];
  const float* V    = (const float*)d_in[2];
  const float* mask = (const float*)d_in[3];
  const float* Wq   = (const float*)d_in[4];
  const float* bq   = (const float*)d_in[5];
  const float* Wk   = (const float*)d_in[6];
  const float* bk   = (const float*)d_in[7];
  const float* Wv   = (const float*)d_in[8];
  const float* bv   = (const float*)d_in[9];
  const float* Wo   = (const float*)d_in[10];
  const float* bo   = (const float*)d_in[11];

  float* out  = (float*)d_out;
  float* attn = out + (size_t)BATCH * SEQ * DMODEL;   // [B,H,S,S] after out

  const size_t n_qkv = (size_t)BATCH * SEQ * DMODEL;  // 8M elements each
  __bf16* qb  = (__bf16*)d_ws;
  __bf16* kb  = qb + n_qkv;
  __bf16* vb  = kb + n_qkv;
  __bf16* ctx = vb + n_qkv;                           // 64 MB total bf16

  dim3 blk(256);
  dim3 gproj(DMODEL / 128, (BATCH * SEQ) / 128);      // (8, 64)

  gemm_bias_kernel<false, false><<<gproj, blk, 0, stream>>>(Q, Wq, bq, qb);
  gemm_bias_kernel<false, false><<<gproj, blk, 0, stream>>>(K, Wk, bk, kb);
  gemm_bias_kernel<false, false><<<gproj, blk, 0, stream>>>(V, Wv, bv, vb);

  // Fused scores + softmax: one block = 64 q-rows x full 1024 k-cols (TDM-fed)
  scores_softmax_kernel<<<dim3(SEQ / 64, BATCH * NHEAD), blk, 0, stream>>>(
      qb, kb, mask, attn);

  attnv_kernel<<<dim3(SEQ / 128, BATCH * NHEAD), blk, 0, stream>>>(attn, vb, ctx);

  gemm_bias_kernel<true, true><<<gproj, blk, 0, stream>>>(ctx, Wo, bo, out);
}
